// CrossAttentionAdapter_89008902242453
// MI455X (gfx1250) — matrix operand order repaired and checked by reading, not compiled, not run on hardware
//
#include <hip/hip_runtime.h>
#include <hip/hip_bf16.h>

typedef __attribute__((ext_vector_type(16))) __bf16 v16bf;
typedef __attribute__((ext_vector_type(8)))  __bf16 v8bf;
typedef __attribute__((ext_vector_type(8)))  float  v8f;
typedef __attribute__((ext_vector_type(4)))  float  v4f;

#define D_      2048
#define H_      16
#define HD_     128
#define INNER_  2048
#define B_      2
#define T_      2048
#define NWS_    64
#define TC_     512
#define NKV_    576            // 64 + 512
#define MQ_     (B_ * T_)      // 4096
#define MKV_    (B_ * NKV_)    // 1152
#define QSCALE_ 0.08838834764831845f   // 1/sqrt(128)

// ---------------------------------------------------------------------------
// helpers
// ---------------------------------------------------------------------------
__device__ __forceinline__ v8f vzero8() {
  v8f v;
#pragma unroll
  for (int i = 0; i < 8; ++i) v[i] = 0.f;
  return v;
}

#define WMMA_BF16(a, b, c) \
  __builtin_amdgcn_wmma_f32_16x16x32_bf16(false, (a), false, (b), (short)0, (c), false, false)

// Async global -> LDS copy, 16 bytes per lane, tracked by ASYNCcnt.
// LDS aperture keeps the wave-relative offset in addr[31:0], so truncating the
// flat address of a __shared__ element yields the LDS byte offset.
__device__ __forceinline__ void async_cp16(const __bf16* gsrc, __bf16* ldst) {
  unsigned loff = (unsigned)(unsigned long long)ldst;
  asm volatile("global_load_async_to_lds_b128 %0, %1, off"
               :: "v"(loff), "v"(gsrc)
               : "memory");
}
__device__ __forceinline__ void wait_async0() {
#if __has_builtin(__builtin_amdgcn_s_wait_asynccnt)
  __builtin_amdgcn_s_wait_asynccnt(0);
#else
  asm volatile("s_wait_asynccnt 0" ::: "memory");
#endif
}

// A fragment (16x32 bf16, MxK). p0 points at element [m=0][k=0] of the tile,
// row-major with leading dim lda (elements).
// lane 0-15 : M=lane,   elems 0-7 -> K=0..7,  elems 8-15 -> K=16..23
// lane 16-31: M=lane-16,elems 0-7 -> K=8..15, elems 8-15 -> K=24..31
template <typename T>
__device__ __forceinline__ v16bf load_frag_a(const T* p0, int lda, int lane) {
  int half = lane >> 4, ml = lane & 15;
  const T* p = p0 + (long)ml * lda + half * 8;
  v8bf lo = *(const v8bf*)p;
  v8bf hi = *(const v8bf*)(p + 16);
  return __builtin_shufflevector(lo, hi, 0, 1, 2, 3, 4, 5, 6, 7, 8, 9, 10, 11, 12, 13, 14, 15);
}

// B fragment (32x16 bf16, KxN) supplied as Bt: N-major (row n holds K values),
// leading dim ldb (elements). p0 points at Bt element [n=0][k=0].
// lane 0-15 : N=lane,    elems 0-7 -> K=0..7,  elems 8-15 -> K=16..23
// lane 16-31: N=lane-16, elems 0-7 -> K=8..15, elems 8-15 -> K=24..31
template <typename T>
__device__ __forceinline__ v16bf load_frag_b(const T* p0, int ldb, int lane) {
  int half = lane >> 4, nl = lane & 15;
  const T* p = p0 + (long)nl * ldb + half * 8;
  v8bf lo = *(const v8bf*)p;
  v8bf hi = *(const v8bf*)(p + 16);
  return __builtin_shufflevector(lo, hi, 0, 1, 2, 3, 4, 5, 6, 7, 8, 9, 10, 11, 12, 13, 14, 15);
}

// ---------------------------------------------------------------------------
// weight transpose + cast: W (K x N f32, K=N=2048) -> Wt (N x K bf16)
// ---------------------------------------------------------------------------
__global__ __launch_bounds__(256) void transpose_cast_kernel(
    const float* __restrict__ W, __bf16* __restrict__ Wt) {
  __shared__ float tile[32][33];
  int tid = threadIdx.x;
  int tx = tid & 31, ty = tid >> 5;  // 32 x 8
  int n0 = blockIdx.x * 32, k0 = blockIdx.y * 32;
#pragma unroll
  for (int i = 0; i < 32; i += 8)
    tile[ty + i][tx] = W[(long)(k0 + ty + i) * D_ + n0 + tx];
  __syncthreads();
#pragma unroll
  for (int i = 0; i < 32; i += 8)
    Wt[(long)(n0 + ty + i) * D_ + k0 + tx] = (__bf16)tile[tx][ty + i];
}

// ---------------------------------------------------------------------------
// v transpose: v (B,NKV,H,HD bf16) -> vt (B,H,HD,NKV bf16)
// ---------------------------------------------------------------------------
__global__ __launch_bounds__(256) void transpose_v_kernel(
    const __bf16* __restrict__ v, __bf16* __restrict__ vt) {
  __shared__ __bf16 tile[32][33];
  int tid = threadIdx.x;
  int tx = tid & 31, ty = tid >> 5;
  int s0 = blockIdx.x * 32;          // NKV/32 = 18
  int d0 = blockIdx.y * 32;          // HD/32  = 4
  int bh = blockIdx.z;               // B*H    = 32
  int b = bh >> 4, h = bh & 15;
#pragma unroll
  for (int i = 0; i < 32; i += 8)
    tile[ty + i][tx] = v[(long)(b * NKV_ + s0 + ty + i) * INNER_ + h * HD_ + d0 + tx];
  __syncthreads();
#pragma unroll
  for (int i = 0; i < 32; i += 8)
    vt[((long)(bh * HD_ + d0 + ty + i)) * NKV_ + s0 + tx] = tile[tx][ty + i];
}

// ---------------------------------------------------------------------------
// RMSNorm (D=2048) -> bf16. One 256-thread block per row (8 elems/thread).
// ---------------------------------------------------------------------------
__device__ __forceinline__ void rmsnorm_row(const float* __restrict__ xr,
                                            const float* __restrict__ w,
                                            __bf16* __restrict__ outr,
                                            float* red, int tid) {
  v4f a = *(const v4f*)(xr + tid * 8);
  v4f b = *(const v4f*)(xr + tid * 8 + 4);
  float s = a[0] * a[0] + a[1] * a[1] + a[2] * a[2] + a[3] * a[3] +
            b[0] * b[0] + b[1] * b[1] + b[2] * b[2] + b[3] * b[3];
  red[tid] = s;
  __syncthreads();
  for (int off = 128; off > 0; off >>= 1) {
    if (tid < off) red[tid] += red[tid + off];
    __syncthreads();
  }
  float inv = rsqrtf(red[0] * (1.0f / (float)D_) + 1e-6f);
  v4f w0 = *(const v4f*)(w + tid * 8);
  v4f w1 = *(const v4f*)(w + tid * 8 + 4);
  v8bf o;
#pragma unroll
  for (int j = 0; j < 4; ++j) o[j] = (__bf16)(a[j] * inv * w0[j]);
#pragma unroll
  for (int j = 0; j < 4; ++j) o[4 + j] = (__bf16)(b[j] * inv * w1[j]);
  *(v8bf*)(outr + tid * 8) = o;
}

__global__ __launch_bounds__(256) void rmsnorm_h_kernel(
    const float* __restrict__ x, const float* __restrict__ w,
    __bf16* __restrict__ out) {
  __shared__ float red[256];
  long row = blockIdx.x;
  rmsnorm_row(x + row * D_, w, out + row * D_, red, threadIdx.x);
}

__global__ __launch_bounds__(256) void rmsnorm_kv_kernel(
    const float* __restrict__ ws, const float* __restrict__ corr,
    const float* __restrict__ w, __bf16* __restrict__ out) {
  __shared__ float red[256];
  long row = blockIdx.x;                 // 0 .. B*NKV-1
  long b = row / NKV_, i = row % NKV_;
  const float* xr = (i < NWS_) ? ws + (b * NWS_ + i) * D_
                               : corr + (b * TC_ + (i - NWS_)) * D_;
  rmsnorm_row(xr, w, out + row * D_, red, threadIdx.x);
}

// ---------------------------------------------------------------------------
// bf16 WMMA GEMM: C(MxN) = A(MxK) * B(KxN), B supplied as Bt (N-major, NxK).
// Block tile 128x128, 8 waves (each 64x32), K-step 64, double-buffered LDS
// filled with global_load_async_to_lds_b128 (ASYNCcnt-tracked), padded stride
// 72 elems -> conflict-free ds_load_b128 fragment reads.
// ---------------------------------------------------------------------------
#define KT_  64
#define ALD2 72

template <int OUTF32>
__global__ __launch_bounds__(256) void gemm_bf16_kernel(
    const __bf16* __restrict__ A, const __bf16* __restrict__ Bt,
    void* __restrict__ Cout, int M, int N, int K, float scale) {
  __shared__ __align__(16) __bf16 As[2][128 * ALD2];
  __shared__ __align__(16) __bf16 Bs[2][128 * ALD2];
  int tid = threadIdx.x;
  int wave = tid >> 5, lane = tid & 31;
  int wm = (wave >> 2) * 64;   // 0 / 64
  int wn = (wave & 3) * 32;    // 0,32,64,96
  long m0 = (long)blockIdx.y * 128;
  long n0 = (long)blockIdx.x * 128;

  int row = tid >> 1;           // 0..127
  int c0 = (tid & 1) * 32;      // element offset within 64-wide K slice

  v8f acc[4][2];
#pragma unroll
  for (int mt = 0; mt < 4; ++mt)
#pragma unroll
    for (int nt = 0; nt < 2; ++nt) acc[mt][nt] = vzero8();

  const __bf16* gA = A + (m0 + row) * (long)K + c0;
  const __bf16* gB = Bt + (n0 + row) * (long)K + c0;

  auto stage = [&](int buf, int k0) {
    __bf16* la = &As[buf][row * ALD2 + c0];
    __bf16* lb = &Bs[buf][row * ALD2 + c0];
    const __bf16* pa = gA + k0;
    const __bf16* pb = gB + k0;
#pragma unroll
    for (int c = 0; c < 32; c += 8) {
      async_cp16(pa + c, la + c);
      async_cp16(pb + c, lb + c);
    }
  };

  auto compute = [&](int buf) {
#pragma unroll
    for (int kk = 0; kk < KT_; kk += 32) {
      v16bf bfrag[2];
#pragma unroll
      for (int nt = 0; nt < 2; ++nt)
        bfrag[nt] = load_frag_b(&Bs[buf][(wn + nt * 16) * ALD2 + kk], ALD2, lane);
#pragma unroll
      for (int mt = 0; mt < 4; ++mt) {
        v16bf afrag = load_frag_a(&As[buf][(wm + mt * 16) * ALD2 + kk], ALD2, lane);
#pragma unroll
        for (int nt = 0; nt < 2; ++nt)
          acc[mt][nt] = WMMA_BF16(afrag, bfrag[nt], acc[mt][nt]);
      }
    }
  };

  stage(0, 0);
  wait_async0();
  __syncthreads();
  int nit = K / KT_;
  for (int it = 0; it < nit; ++it) {
    int cur = it & 1;
    if (it + 1 < nit) stage(cur ^ 1, (it + 1) * KT_);
    compute(cur);
    if (it + 1 < nit) wait_async0();
    __syncthreads();
  }

  int nl = lane & 15, half = lane >> 4;
#pragma unroll
  for (int mt = 0; mt < 4; ++mt)
#pragma unroll
    for (int nt = 0; nt < 2; ++nt) {
      long ncol = n0 + wn + nt * 16 + nl;
#pragma unroll
      for (int r = 0; r < 8; ++r) {
        long mrow = m0 + wm + mt * 16 + half * 8 + r;
        float val = acc[mt][nt][r] * scale;
        if (OUTF32)
          ((float*)Cout)[mrow * N + ncol] = val;
        else
          ((__bf16*)Cout)[mrow * N + ncol] = (__bf16)val;
      }
    }
}

// ---------------------------------------------------------------------------
// Attention: one 4-wave block per (b, h, 16-query tile).
// scores (16 x 576) via WMMA -> LDS f32 (+ precomputed additive mask bias);
// masked softmax (normalization deferred to PV epilogue); probs bf16 in LDS;
// PV via WMMA against pre-transposed Vt.
// ---------------------------------------------------------------------------
#define SLD 592   // padded LDS leading dim (elements); 592*4 and 592*2 are 16B multiples

__global__ __launch_bounds__(128) void attn_kernel(
    const __bf16* __restrict__ q, const __bf16* __restrict__ k,
    const __bf16* __restrict__ vt, const int* __restrict__ cmask,
    __bf16* __restrict__ ao) {
  __shared__ __align__(16) float sc[16 * SLD];
  __shared__ __align__(16) __bf16 pr[16 * SLD];
  __shared__ float bias[NKV_];
  __shared__ float red_max[16][9];
  __shared__ float red_sum[16][9];
  __shared__ float rinv[16];

  int tid = threadIdx.x;
  int wave = tid >> 5, lane = tid & 31;
  int bid = blockIdx.x;           // B*H*(T/16) = 4096
  int qt = bid & 127;
  int h = (bid >> 7) & 15;
  int b = bid >> 11;

  // additive mask bias, once per block (keeps the WMMA loop branch-free)
  for (int c = tid; c < NKV_; c += 128) {
    bool valid = (c < NWS_) || (cmask[b * TC_ + (c - NWS_)] != 0);
    bias[c] = valid ? 0.f : -1e30f;
  }

  const __bf16* qp = q + ((long)(b * T_ + qt * 16)) * INNER_ + h * HD_;  // lda = INNER_
  const __bf16* kp = k + (long)b * NKV_ * INNER_ + h * HD_;              // ldb = INNER_

  // q fragments for the whole HD=128 reduction (4 k-steps), shared by all tiles
  v16bf qfrag[4];
#pragma unroll
  for (int kk = 0; kk < 4; ++kk)
    qfrag[kk] = load_frag_a(qp + kk * 32, INNER_, lane);
  __syncthreads();   // bias visible

  // ---- scores = (q * scale) @ k^T + bias, into LDS ----
  for (int ct = wave; ct < NKV_ / 16; ct += 4) {
    int s0 = ct * 16;
    v16bf bb[4];
#pragma unroll
    for (int kk = 0; kk < 4; ++kk)
      bb[kk] = load_frag_b(kp + (long)s0 * INNER_ + kk * 32, INNER_, lane);
    v8f acc = vzero8();
#pragma unroll
    for (int kk = 0; kk < 4; ++kk)
      acc = WMMA_BF16(qfrag[kk], bb[kk], acc);
    int nl = lane & 15, half = lane >> 4;
    int col = s0 + nl;
    float bcol = bias[col];
#pragma unroll
    for (int r = 0; r < 8; ++r)
      sc[(half * 8 + r) * SLD + col] = acc[r] + bcol;
  }
  __syncthreads();

  // ---- softmax (deferred normalization): 16 rows x 8 segments of 72 ----
  int rrow = tid & 15, seg = tid >> 4;
  float mx = -1e30f;
  for (int c = seg * 72; c < seg * 72 + 72; ++c)
    mx = fmaxf(mx, sc[rrow * SLD + c]);
  red_max[rrow][seg] = mx;
  __syncthreads();
  mx = red_max[rrow][0];
#pragma unroll
  for (int i = 1; i < 8; ++i) mx = fmaxf(mx, red_max[rrow][i]);
  float sum = 0.f;
  for (int c = seg * 72; c < seg * 72 + 72; ++c) {
    float e = __expf(sc[rrow * SLD + c] - mx);
    pr[rrow * SLD + c] = (__bf16)e;
    sum += e;
  }
  red_sum[rrow][seg] = sum;
  __syncthreads();
  if (tid < 16) {
    float s = 0.f;
#pragma unroll
    for (int i = 0; i < 8; ++i) s += red_sum[tid][i];
    rinv[tid] = 1.0f / s;
  }
  __syncthreads();

  // ---- out = P @ V : each wave computes 16 x 32 of the 16 x 128 result ----
  const __bf16* vtp = vt + ((long)(b * H_ + h) * HD_) * NKV_;  // row hd, stride NKV_
  v8f o0 = vzero8(), o1 = vzero8();
  for (int k0 = 0; k0 < NKV_; k0 += 32) {
    v16bf a = load_frag_a(&pr[k0], SLD, lane);
    v16bf b0 = load_frag_b(vtp + (long)(wave * 32) * NKV_ + k0, NKV_, lane);
    v16bf b1 = load_frag_b(vtp + (long)(wave * 32 + 16) * NKV_ + k0, NKV_, lane);
    o0 = WMMA_BF16(a, b0, o0);
    o1 = WMMA_BF16(a, b1, o1);
  }
  int nl = lane & 15, half = lane >> 4;
#pragma unroll
  for (int nt = 0; nt < 2; ++nt) {
    v8f& oc = nt ? o1 : o0;
    int hd = wave * 32 + nt * 16 + nl;
#pragma unroll
    for (int r = 0; r < 8; ++r) {
      int m = half * 8 + r;
      int t = qt * 16 + m;
      float val = oc[r] * rinv[m];
      ao[((long)(b * T_ + t)) * INNER_ + h * HD_ + hd] = (__bf16)val;
    }
  }
}

// ---------------------------------------------------------------------------
// launch
// ---------------------------------------------------------------------------
extern "C" void kernel_launch(void* const* d_in, const int* in_sizes, int n_in,
                              void* d_out, int out_size, void* d_ws, size_t ws_size,
                              hipStream_t stream) {
  const float* hs   = (const float*)d_in[0];   // (B,T,D)
  const float* wsp  = (const float*)d_in[1];   // (B,NWS,D)
  const float* corr = (const float*)d_in[2];   // (B,TC,D)
  const int*  cmask = (const int*)d_in[3];     // (B,TC)
  const float* lnq  = (const float*)d_in[4];
  const float* lnkv = (const float*)d_in[5];
  const float* Wq   = (const float*)d_in[6];
  const float* Wk   = (const float*)d_in[7];
  const float* Wv   = (const float*)d_in[8];
  const float* Wo   = (const float*)d_in[9];
  float* out = (float*)d_out;

  const size_t WBYTES  = (size_t)D_ * INNER_ * 2;       // 8 MB each (bf16)
  const size_t HBYTES  = (size_t)MQ_ * D_ * 2;          // 16 MB
  const size_t KVBYTES = (size_t)MKV_ * D_ * 2;         // 4.5 MB

  char* p = (char*)d_ws;
  __bf16* Wq_t = (__bf16*)p;  p += WBYTES;
  __bf16* Wk_t = (__bf16*)p;  p += WBYTES;
  __bf16* Wv_t = (__bf16*)p;  p += WBYTES;
  __bf16* Wo_t = (__bf16*)p;  p += WBYTES;
  __bf16* h_bf = (__bf16*)p;  p += HBYTES;
  __bf16* kv_bf = (__bf16*)p; p += KVBYTES;
  __bf16* q_bf = (__bf16*)p;  p += HBYTES;
  __bf16* k_bf = (__bf16*)p;  p += KVBYTES;
  __bf16* v_bf = (__bf16*)p;  p += KVBYTES;
  __bf16* vt_bf = (__bf16*)p; p += KVBYTES;
  __bf16* ao_bf = (__bf16*)p; p += HBYTES;

  dim3 tgrid(D_ / 32, D_ / 32);
  transpose_cast_kernel<<<tgrid, 256, 0, stream>>>(Wq, Wq_t);
  transpose_cast_kernel<<<tgrid, 256, 0, stream>>>(Wk, Wk_t);
  transpose_cast_kernel<<<tgrid, 256, 0, stream>>>(Wv, Wv_t);
  transpose_cast_kernel<<<tgrid, 256, 0, stream>>>(Wo, Wo_t);

  rmsnorm_h_kernel<<<MQ_, 256, 0, stream>>>(hs, lnq, h_bf);
  rmsnorm_kv_kernel<<<MKV_, 256, 0, stream>>>(wsp, corr, lnkv, kv_bf);

  // q = h @ Wq  (scale folded in), k = kv @ Wk, v = kv @ Wv
  gemm_bf16_kernel<0><<<dim3(INNER_ / 128, MQ_ / 128), 256, 0, stream>>>(
      h_bf, Wq_t, q_bf, MQ_, INNER_, D_, QSCALE_);
  gemm_bf16_kernel<0><<<dim3(INNER_ / 128, MKV_ / 128), 256, 0, stream>>>(
      kv_bf, Wk_t, k_bf, MKV_, INNER_, D_, 1.0f);
  gemm_bf16_kernel<0><<<dim3(INNER_ / 128, MKV_ / 128), 256, 0, stream>>>(
      kv_bf, Wv_t, v_bf, MKV_, INNER_, D_, 1.0f);

  transpose_v_kernel<<<dim3(NKV_ / 32, HD_ / 32, B_ * H_), 256, 0, stream>>>(v_bf, vt_bf);

  attn_kernel<<<B_ * H_ * (T_ / 16), 128, 0, stream>>>(q_bf, k_bf, vt_bf, cmask, ao_bf);

  // out = ao @ Wo  (f32 output)
  gemm_bf16_kernel<1><<<dim3(D_ / 128, MQ_ / 128), 256, 0, stream>>>(
      ao_bf, Wo_t, out, MQ_, D_, INNER_, 1.0f);
}
